// Attention_54245436948569
// MI455X (gfx1250) — matrix operand order repaired and checked by reading, not compiled, not run on hardware
//
#include <hip/hip_runtime.h>

#define DEV __device__ __forceinline__

typedef __attribute__((ext_vector_type(16))) __bf16       v16bf;
typedef __attribute__((ext_vector_type(2)))  __bf16       v2bf;
typedef __attribute__((ext_vector_type(8)))  float        v8f;
typedef __attribute__((ext_vector_type(4)))  unsigned int v4u;
typedef int v4i_gcc __attribute__((vector_size(16)));     // matches builtin param

#if __has_builtin(__builtin_amdgcn_global_load_async_to_lds_b128) && \
    __has_builtin(__builtin_amdgcn_s_wait_asynccnt)
#define USE_ASYNC_LDS 1
#else
#define USE_ASYNC_LDS 0
#endif

union Frag {
  v16bf v;
  v4u   q[2];
};

// round-to-nearest-even f32 -> bf16 (raw bits)
DEV unsigned short f2bf(float f) {
  unsigned int u = __float_as_uint(f);
  u += 0x7FFFu + ((u >> 16) & 1u);
  return (unsigned short)(u >> 16);
}

// two f32 -> packed bf16 pair (RNE); hw pack instruction when available
DEV unsigned int pack2(float lo, float hi) {
#if __has_builtin(__builtin_amdgcn_cvt_pk_bf16_f32)
  v2bf p = __builtin_amdgcn_cvt_pk_bf16_f32(lo, hi);
  return __builtin_bit_cast(unsigned int, p);
#else
  return (unsigned int)f2bf(lo) | ((unsigned int)f2bf(hi) << 16);
#endif
}

// 16-byte global -> LDS copy. Async (ASYNCcnt-tracked, overlaps compute) when
// the gfx1250 async-to-LDS builtins exist; plain b128 VGPR copy otherwise.
DEV void cp16(void* lds_dst, const void* gsrc) {
#if USE_ASYNC_LDS
  __builtin_amdgcn_global_load_async_to_lds_b128(
      (__attribute__((address_space(1))) v4i_gcc*)gsrc,
      (__attribute__((address_space(3))) v4i_gcc*)lds_dst, 0, 0);
#else
  *(v4u*)lds_dst = *(const v4u*)gsrc;
#endif
}

DEV void cp_wait_all() {
#if USE_ASYNC_LDS
  __builtin_amdgcn_s_wait_asynccnt(0);
#endif
}

// A-fragment, 16x32 bf16 tile row-major in LDS (stride in halfs, mult of 8).
// Lane l: row = l&15; VGPRs0-3: K=(l>>4)*8..+7 ; VGPRs4-7: K=16+(l>>4)*8..+7
DEV v16bf load_fragA(const unsigned short* p, int stride_h) {
  const int lane = threadIdx.x & 31;
  const unsigned short* b = p + (lane & 15) * stride_h + (lane >> 4) * 8;
  Frag f;
  f.q[0] = *(const v4u*)(b);
  f.q[1] = *(const v4u*)(b + 16);
  return f.v;
}

// B-fragment, 32x16 bf16 stored transposed in LDS as [n][k] row-major.
// Lane l: col n = l&15; VGPRs0-3: K=(l>>4)*8..+7 ; VGPRs4-7: K=16+(l>>4)*8..+7
DEV v16bf load_fragB(const unsigned short* p, int stride_h) {
  const int lane = threadIdx.x & 31;
  const unsigned short* b = p + (lane & 15) * stride_h + (lane >> 4) * 8;
  Frag f;
  f.q[0] = *(const v4u*)(b);
  f.q[1] = *(const v4u*)(b + 16);
  return f.v;
}

DEV v8f wmma_bf16(v16bf a, v16bf b, v8f c) {
  return __builtin_amdgcn_wmma_f32_16x16x32_bf16(false, a, false, b, (short)0, c,
                                                 false, false);
}

// ---------------------------------------------------------------------------
// Kernel 1: QKV = X @ Wqkv, fused interleaved RoPE on Q,K.
//   X [4096,1024] f32, Wqkv [1024,3072] f32
//   -> Qb,Kb bf16 [B,H,N,DH] ; Vt bf16 [B,H,DH,N] (transposed for PV gemm)
// Block tile 64(M) x 128(N), 8 waves (4x2), each wave 16x64 out.
// ---------------------------------------------------------------------------
__global__ __launch_bounds__(256)
void qkv_rope_kernel(const float* __restrict__ X, const float* __restrict__ W,
                     unsigned short* __restrict__ Qb, unsigned short* __restrict__ Kb,
                     unsigned short* __restrict__ Vt) {
  __shared__ __align__(16) unsigned short lA[64 * 32];
  __shared__ __align__(16) unsigned short lB[128 * 32];
  const int t = threadIdx.x;
  const int lane = t & 31, w = t >> 5;
  const int wm = w >> 1, wn = w & 1;
  const int m0 = blockIdx.x * 64;
  const int n0 = blockIdx.y * 128;

  v8f acc[4];
  #pragma unroll
  for (int j = 0; j < 4; ++j)
    #pragma unroll
    for (int r = 0; r < 8; ++r) acc[j][r] = 0.f;

  for (int k0 = 0; k0 < 1024; k0 += 32) {
    unsigned int* dA = (unsigned int*)lA;
    #pragma unroll
    for (int i = 0; i < 4; ++i) {            // A: 64x32 f32 -> packed bf16 pairs
      int idx = t + i * 256;                 // 1024 dword cells = 64 x 16
      int r = idx >> 4, c2 = idx & 15;
      float2 fv = *(const float2*)&X[(size_t)(m0 + r) * 1024 + k0 + c2 * 2];
      dA[r * 16 + c2] = pack2(fv.x, fv.y);
    }
    unsigned int* dB = (unsigned int*)lB;
    #pragma unroll
    for (int i = 0; i < 8; ++i) {            // B: W[k][n] -> LDS [n][k] packed
      int idx = t + i * 256;                 // 2048 cells = 16 k-pairs x 128 n
      int k2 = idx >> 7, n = idx & 127;
      float lo = W[(size_t)(k0 + 2 * k2) * 3072 + n0 + n];
      float hi = W[(size_t)(k0 + 2 * k2 + 1) * 3072 + n0 + n];
      dB[n * 16 + k2] = pack2(lo, hi);
    }
    __syncthreads();
    v16bf a = load_fragA(lA + wm * 16 * 32, 32);
    v16bf bf[4];
    #pragma unroll
    for (int j = 0; j < 4; ++j)
      bf[j] = load_fragB(lB + (wn * 64 + j * 16) * 32, 32);
    #pragma unroll
    for (int j = 0; j < 4; ++j)
      acc[j] = wmma_bf16(a, bf[j], acc[j]);
    __syncthreads();
  }

  // Epilogue. C layout: lane l holds (m = 8*(l>>4)+r, n = l&15).
  const int half = lane >> 4, nl = lane & 15;
  const int bb = m0 >> 11;                      // batch (blocks never straddle)
  const int posb = (m0 & 2047) + wm * 16 + half * 8;
  #pragma unroll
  for (int j = 0; j < 4; ++j) {
    const int c = n0 + wn * 64 + j * 16 + nl;   // 0..3071
    const int region = c >> 10;                 // 0=Q 1=K 2=V (wave-uniform)
    const int ch = c & 1023;
    const int hh = ch >> 6, d = ch & 63;
    if (region < 2) {
      unsigned short* dst = (region == 0) ? Qb : Kb;
      const float invf = __powf(10000.0f, -(float)(d & ~1) / 64.0f);
      #pragma unroll
      for (int r = 0; r < 8; ++r) {
        const int pos = posb + r;
        float v = acc[j][r];
        float pv = __shfl_xor(v, 1, 32);        // RoPE pair partner (adjacent n)
        float sn, cs;
        __sincosf((float)pos * invf, &sn, &cs);
        float rv = (d & 1) ? fmaf(pv, sn, v * cs) : fmaf(-pv, sn, v * cs);
        float rhi = __shfl_xor(rv, 1, 32);      // neighbor's rotated value
        if (!(d & 1)) {                         // even lane stores packed pair
          size_t o = (((size_t)(bb * 16 + hh) * 2048 + pos) * 64 + d);
          *(unsigned int*)(dst + o) = pack2(rv, rhi);
        }
      }
    } else {
      const size_t vb = ((size_t)((bb * 16 + hh) * 64 + d)) * 2048 + posb;
      #pragma unroll
      for (int r = 0; r < 8; r += 2)            // pack along pos (in-lane)
        *(unsigned int*)(Vt + vb + r) = pack2(acc[j][r], acc[j][r + 1]);
    }
  }
}

// ---------------------------------------------------------------------------
// Kernel 2: flash attention per (b,h). Block = 128 query rows, 8 waves x 16 rows.
// Key blocks of 64, double-buffered in LDS via async-to-LDS copies.
// ---------------------------------------------------------------------------
__global__ __launch_bounds__(256)
void attn_kernel(const unsigned short* __restrict__ Qb,
                 const unsigned short* __restrict__ Kb,
                 const unsigned short* __restrict__ Vt,
                 unsigned short* __restrict__ Ob) {
  __shared__ __align__(16) unsigned short lQ[128 * 64];     // 16 KB
  __shared__ __align__(16) unsigned short lK[2][64 * 64];   // 2 x 8 KB
  __shared__ __align__(16) unsigned short lV[2][64 * 64];   // 2 x 8 KB [dh][key]
  __shared__ __align__(16) unsigned short lP[8 * 16 * 64];  // 16 KB per-wave P
  const int t = threadIdx.x, lane = t & 31, w = t >> 5;
  const int bh = blockIdx.x;            // 0..31
  const int q0 = blockIdx.y * 128;
  const unsigned short* Qg = Qb + (size_t)bh * 2048 * 64;
  const unsigned short* Kg = Kb + (size_t)bh * 2048 * 64;
  const unsigned short* Vg = Vt + (size_t)bh * 64 * 2048;

  auto stage_kv = [&](int key0, int buf) {
    #pragma unroll
    for (int i = 0; i < 2; ++i) {
      int chunk = t + i * 256;                     // 512 chunks x 8 halfs
      cp16(&lK[buf][chunk * 8], Kg + (size_t)key0 * 64 + chunk * 8);
      int rr = chunk >> 3, cc = chunk & 7;         // dh row, key chunk
      cp16(&lV[buf][rr * 64 + cc * 8], Vg + (size_t)rr * 2048 + key0 + cc * 8);
    }
  };

  // prologue: stage Q tile + first K/V block
  #pragma unroll
  for (int i = 0; i < 4; ++i) {
    int chunk = t + i * 256;                       // 1024 chunks x 8 halfs
    cp16(&lQ[chunk * 8], Qg + (size_t)q0 * 64 + chunk * 8);
  }
  stage_kv(0, 0);
  cp_wait_all();
  __syncthreads();

  v16bf qf[2];
  qf[0] = load_fragA(lQ + w * 16 * 64, 64);
  qf[1] = load_fragA(lQ + w * 16 * 64 + 32, 64);

  float m_s[8], l_s[8];
  v8f acc[4];
  #pragma unroll
  for (int r = 0; r < 8; ++r) { m_s[r] = -3.0e38f; l_s[r] = 0.f; }
  #pragma unroll
  for (int j = 0; j < 4; ++j)
    #pragma unroll
    for (int r = 0; r < 8; ++r) acc[j][r] = 0.f;

  for (int kb = 0; kb < 32; ++kb) {
    const int cur = kb & 1;
    if (kb + 1 < 32) stage_kv((kb + 1) * 64, cur ^ 1);  // overlap with compute
    const unsigned short* lKc = lK[cur];
    const unsigned short* lVc = lV[cur];

    // S = (Q @ K^T) * scale
    v8f S[4];
    #pragma unroll
    for (int j = 0; j < 4; ++j) {
      v16bf b0 = load_fragB(lKc + (j * 16) * 64, 64);
      v16bf b1 = load_fragB(lKc + (j * 16) * 64 + 32, 64);
      v8f s;
      #pragma unroll
      for (int r = 0; r < 8; ++r) s[r] = 0.f;
      s = wmma_bf16(qf[0], b0, s);
      s = wmma_bf16(qf[1], b1, s);
      S[j] = s;
    }

    // online softmax (row stats across the 16 lanes of each half-group)
    #pragma unroll
    for (int r = 0; r < 8; ++r) {
      float mx = m_s[r];
      #pragma unroll
      for (int j = 0; j < 4; ++j) {
        float v = S[j][r] * 0.125f;               // 1/sqrt(64)
        S[j][r] = v;
        mx = fmaxf(mx, v);
      }
      #pragma unroll
      for (int off = 1; off < 16; off <<= 1) mx = fmaxf(mx, __shfl_xor(mx, off, 32));
      const float corr = __expf(m_s[r] - mx);
      float rs = 0.f;
      #pragma unroll
      for (int j = 0; j < 4; ++j) {
        float p = __expf(S[j][r] - mx);
        S[j][r] = p;
        rs += p;
      }
      #pragma unroll
      for (int off = 1; off < 16; off <<= 1) rs += __shfl_xor(rs, off, 32);
      l_s[r] = l_s[r] * corr + rs;
      m_s[r] = mx;
      #pragma unroll
      for (int j = 0; j < 4; ++j) acc[j][r] *= corr;
    }

    // P -> wave-private LDS (C layout -> A-fragment relayout)
    {
      unsigned short* pw = lP + w * 1024;
      const int half = lane >> 4, nl = lane & 15;
      #pragma unroll
      for (int j = 0; j < 4; ++j)
        #pragma unroll
        for (int r = 0; r < 8; ++r)
          pw[(half * 8 + r) * 64 + j * 16 + nl] = f2bf(S[j][r]);
    }
    __builtin_amdgcn_wave_barrier();

    // acc += P @ V  (B = V^T tile already [dh][key] in LDS)
    v16bf pa[2];
    pa[0] = load_fragA(lP + w * 1024, 64);
    pa[1] = load_fragA(lP + w * 1024 + 32, 64);
    #pragma unroll
    for (int j = 0; j < 4; ++j) {
      v16bf b0 = load_fragB(lVc + (j * 16) * 64, 64);
      v16bf b1 = load_fragB(lVc + (j * 16) * 64 + 32, 64);
      acc[j] = wmma_bf16(pa[0], b0, acc[j]);
      acc[j] = wmma_bf16(pa[1], b1, acc[j]);
    }

    cp_wait_all();          // next buffer's async loads landed
    __syncthreads();
  }

  // normalize + store O as [B*N][H*DH] bf16
  const int b = bh >> 4, h = bh & 15;
  const int half = lane >> 4, nl = lane & 15;
  #pragma unroll
  for (int r = 0; r < 8; ++r) {
    const float inv = 1.0f / l_s[r];
    const int pos = q0 + w * 16 + half * 8 + r;
    const size_t base = ((size_t)(b * 2048 + pos)) * 1024 + h * 64;
    #pragma unroll
    for (int j = 0; j < 4; ++j)
      Ob[base + j * 16 + nl] = f2bf(acc[j][r] * inv);
  }
}

// ---------------------------------------------------------------------------
// Kernel 3: out = O @ Wout + bias.  O bf16 [4096,1024], Wout f32 [1024,1024].
// ---------------------------------------------------------------------------
__global__ __launch_bounds__(256)
void proj_kernel(const unsigned short* __restrict__ Ob, const float* __restrict__ W,
                 const float* __restrict__ bias, float* __restrict__ out) {
  __shared__ __align__(16) unsigned short lA[64 * 32];
  __shared__ __align__(16) unsigned short lB[128 * 32];
  const int t = threadIdx.x, lane = t & 31, w = t >> 5;
  const int wm = w >> 1, wn = w & 1;
  const int m0 = blockIdx.x * 64, n0 = blockIdx.y * 128;

  v8f acc[4];
  #pragma unroll
  for (int j = 0; j < 4; ++j)
    #pragma unroll
    for (int r = 0; r < 8; ++r) acc[j][r] = 0.f;

  for (int k0 = 0; k0 < 1024; k0 += 32) {
    {
      // A already bf16: one b128 (async-capable) chunk per thread
      int r = t >> 2, c4 = t & 3;
      cp16(&lA[r * 32 + c4 * 8], Ob + (size_t)(m0 + r) * 1024 + k0 + c4 * 8);
    }
    unsigned int* dB = (unsigned int*)lB;
    #pragma unroll
    for (int i = 0; i < 8; ++i) {                  // Wout f32 -> bf16 [n][k] packed
      int idx = t + i * 256;                       // 2048 cells = 16 k-pairs x 128 n
      int k2 = idx >> 7, n = idx & 127;
      float lo = W[(size_t)(k0 + 2 * k2) * 1024 + n0 + n];
      float hi = W[(size_t)(k0 + 2 * k2 + 1) * 1024 + n0 + n];
      dB[n * 16 + k2] = pack2(lo, hi);
    }
    cp_wait_all();
    __syncthreads();
    v16bf a = load_fragA(lA + wm * 16 * 32, 32);
    v16bf bf[4];
    #pragma unroll
    for (int j = 0; j < 4; ++j)
      bf[j] = load_fragB(lB + (wn * 64 + j * 16) * 32, 32);
    #pragma unroll
    for (int j = 0; j < 4; ++j)
      acc[j] = wmma_bf16(a, bf[j], acc[j]);
    __syncthreads();
  }

  const int half = lane >> 4, nl = lane & 15;
  #pragma unroll
  for (int j = 0; j < 4; ++j) {
    const int c = n0 + wn * 64 + j * 16 + nl;
    const float bv = bias[c];
    #pragma unroll
    for (int r = 0; r < 8; ++r) {
      const int gm = m0 + wm * 16 + half * 8 + r;
      out[(size_t)gm * 1024 + c] = acc[j][r] + bv;
    }
  }
}

// ---------------------------------------------------------------------------
extern "C" void kernel_launch(void* const* d_in, const int* in_sizes, int n_in,
                              void* d_out, int out_size, void* d_ws, size_t ws_size,
                              hipStream_t stream) {
  const float* X    = (const float*)d_in[0];   // [2,2048,1024]
  const float* Wqkv = (const float*)d_in[1];   // [1024,3072]
  const float* Wout = (const float*)d_in[2];   // [1024,1024]
  const float* bout = (const float*)d_in[3];   // [1024]
  float* out = (float*)d_out;                  // [2,2048,1024]

  unsigned short* ws = (unsigned short*)d_ws;
  const size_t seg = (size_t)2 * 16 * 2048 * 64;   // 4M bf16 elems = 8 MB
  unsigned short* Qb = ws;
  unsigned short* Kb = ws + seg;
  unsigned short* Vt = ws + 2 * seg;
  unsigned short* Ob = ws + 3 * seg;

  qkv_rope_kernel<<<dim3(64, 24), 256, 0, stream>>>(X, Wqkv, Qb, Kb, Vt);
  attn_kernel<<<dim3(32, 16), 256, 0, stream>>>(Qb, Kb, Vt, Ob);
  proj_kernel<<<dim3(64, 8), 256, 0, stream>>>(Ob, Wout, bout, out);
}
